// DecoderLayer_3143916061036
// MI455X (gfx1250) — matrix operand order repaired and checked by reading, not compiled, not run on hardware
//
#include <hip/hip_runtime.h>
#include <hip/hip_bf16.h>
#include <math.h>

#define B_ 1024
#define S_ 128
#define H_ 1024
#define F_ 4096
#define KLD_ (H_ + 8)   // padded LDS row stride (elements): 516 dwords -> conflict-free

typedef __attribute__((ext_vector_type(16))) __bf16 v16bf;
typedef __attribute__((ext_vector_type(8)))  __bf16 v8bf;
typedef __attribute__((ext_vector_type(8)))  float  v8f;

__device__ __forceinline__ float sigf(float x) { return 1.0f / (1.0f + __expf(-x)); }

// ---- bf16-native WMMA fragment loaders --------------------------------------
// A (16x32 bf16, row-major, ld=K): lane<16 holds K=0..15 of row
// (lane&15); lane>=16 holds K=16..31.  Two 16B loads, no conversion.
__device__ __forceinline__ v16bf load_fragA_bf(const __bf16* base, int lane) {
    const __bf16* r = base + ((lane & 16) ? 16 : 0);
    v8bf lo = *(const v8bf*)(r);
    v8bf hi = *(const v8bf*)(r + 8);
    v16bf v;
#pragma unroll
    for (int t = 0; t < 8; ++t) { v[t] = lo[t]; v[t + 8] = hi[t]; }
    return v;
}
// B (32x16 bf16) from weight row n (= output column), contiguous along K:
// lane<16 holds K=0..15 of column (lane&15); lane>=16 holds K=16..31. One 32B load.
__device__ __forceinline__ v16bf load_fragB_bf(const __bf16* base, int lane) {
    return *(const v16bf*)(base + ((lane & 16) ? 16 : 0));
}

// ---- one-pass f32 -> bf16 converter ----------------------------------------
__global__ __launch_bounds__(256)
void cvt_bf16(const float* __restrict__ x, __bf16* __restrict__ y) {
    int i = (blockIdx.x * 256 + threadIdx.x) * 4;
    float4 f = *(const float4*)(x + i);
    y[i]     = (__bf16)f.x;
    y[i + 1] = (__bf16)f.y;
    y[i + 2] = (__bf16)f.z;
    y[i + 3] = (__bf16)f.w;
}

// ---- Generic GEMM: C[M,N] = A[M,K] @ W[N,K]^T (+ optional 2nd pair) ---------
// All operands bf16, accumulate f32 in WMMA.
// EPI 0: v = acc + bias0[n] + bias1[n]
// EPI 1: v = silu(aux[idx]) * acc          (SwiGLU combine)
// EPI 2: v = acc + aux[idx]                (residual add)
template <int EPI>
__global__ __launch_bounds__(256)
void gemm_bf16_wmma(const __bf16* __restrict__ A0, const __bf16* __restrict__ W0,
                    const __bf16* __restrict__ A1, const __bf16* __restrict__ W1,
                    const float* __restrict__ bias0, const float* __restrict__ bias1,
                    const float* __restrict__ aux, float* __restrict__ out,
                    __bf16* __restrict__ out_bf, int M, int N, int K)
{
    const int lane = threadIdx.x & 31;
    const int wave = threadIdx.x >> 5;
    const int wm = wave >> 2;                 // 0..1
    const int wn = wave & 3;                  // 0..3
    const int tileM = blockIdx.y * 128 + wm * 64;
    const int tileN = blockIdx.x * 128 + wn * 32;
    const int lm = lane & 15;

    v8f zero = {};
    v8f acc[4][2];
#pragma unroll
    for (int i = 0; i < 4; ++i)
#pragma unroll
        for (int j = 0; j < 2; ++j) acc[i][j] = zero;

    for (int pass = 0; pass < 2; ++pass) {
        const __bf16* A = pass ? A1 : A0;
        const __bf16* W = pass ? W1 : W0;
        if (A == nullptr) break;
        for (int k = 0; k < K; k += 32) {
            v16bf a[4], bb[2];
#pragma unroll
            for (int mf = 0; mf < 4; ++mf) {
                const __bf16* p = A + (size_t)(tileM + mf * 16 + lm) * K + k;
                if (mf == 0) __builtin_prefetch(p + 64, 0, 1);
                a[mf] = load_fragA_bf(p, lane);
            }
#pragma unroll
            for (int nf = 0; nf < 2; ++nf) {
                const __bf16* p = W + (size_t)(tileN + nf * 16 + lm) * K + k;
                if (nf == 0) __builtin_prefetch(p + 64, 0, 1);
                bb[nf] = load_fragB_bf(p, lane);
            }
#pragma unroll
            for (int mf = 0; mf < 4; ++mf)
#pragma unroll
                for (int nf = 0; nf < 2; ++nf)
                    acc[mf][nf] = __builtin_amdgcn_wmma_f32_16x16x32_bf16(
                        false, a[mf], false, bb[nf], (short)0, acc[mf][nf], false, false);
        }
    }

#pragma unroll
    for (int mf = 0; mf < 4; ++mf) {
#pragma unroll
        for (int nf = 0; nf < 2; ++nf) {
#pragma unroll
            for (int r = 0; r < 8; ++r) {
                int row = tileM + mf * 16 + ((lane & 16) ? (8 + r) : r);
                int col = tileN + nf * 16 + lm;
                size_t idx = (size_t)row * N + col;
                float v = acc[mf][nf][r];
                if (EPI == 0) {
                    if (bias0) v += bias0[col];
                    if (bias1) v += bias1[col];
                } else if (EPI == 1) {
                    float g = aux[idx];
                    v = (g * sigf(g)) * v;
                } else {
                    v += aux[idx];
                }
                if (out)    out[idx]    = v;
                if (out_bf) out_bf[idx] = (__bf16)v;
            }
        }
    }
}

// ---- LN1 over rows of x, emits bf16 directly --------------------------------
__global__ __launch_bounds__(256)
void ln_rows_bf(const float* __restrict__ x, const float* __restrict__ g,
                const float* __restrict__ b, __bf16* __restrict__ y_bf)
{
    __shared__ float red[256];
    const int bid = blockIdx.x, tid = threadIdx.x;
    const float* row = x + (size_t)bid * H_;
    float v[4], s = 0.f;
#pragma unroll
    for (int j = 0; j < 4; ++j) { v[j] = row[tid + j * 256]; s += v[j]; }
    red[tid] = s; __syncthreads();
    for (int off = 128; off > 0; off >>= 1) { if (tid < off) red[tid] += red[tid + off]; __syncthreads(); }
    const float mean = red[0] * (1.0f / H_);
    __syncthreads();
    float s2 = 0.f;
#pragma unroll
    for (int j = 0; j < 4; ++j) { float d = v[j] - mean; s2 += d * d; }
    red[tid] = s2; __syncthreads();
    for (int off = 128; off > 0; off >>= 1) { if (tid < off) red[tid] += red[tid + off]; __syncthreads(); }
    const float rstd = rsqrtf(red[0] * (1.0f / H_) + 1e-5f);
#pragma unroll
    for (int j = 0; j < 4; ++j) {
        int h = tid + j * 256;
        y_bf[(size_t)bid * H_ + h] = (__bf16)((v[j] - mean) * rstd * g[h] + b[h]);
    }
}

// ---- LSTM gate nonlinearities + residual + fused LN2 ------------------------
__global__ __launch_bounds__(256)
void lstm_ln2(const float* __restrict__ gates, const float* __restrict__ x,
              const float* __restrict__ c0, const float* __restrict__ g2,
              const float* __restrict__ b2, float* __restrict__ out_newh,
              float* __restrict__ out_cnew, float* __restrict__ h_norm,
              __bf16* __restrict__ h_norm_bf)
{
    __shared__ float red[256];
    const int b = blockIdx.x, tid = threadIdx.x;
    const float* grow = gates + (size_t)b * (4 * H_);
    float nh[4], s = 0.f;
#pragma unroll
    for (int j = 0; j < 4; ++j) {
        int h = tid + j * 256;
        float gi = grow[h], gf = grow[H_ + h], gg = grow[2 * H_ + h], go = grow[3 * H_ + h];
        float c = sigf(gf) * c0[(size_t)b * H_ + h] + sigf(gi) * tanhf(gg);
        float hc = sigf(go) * tanhf(c);
        float v = x[(size_t)b * H_ + h] + hc;
        out_cnew[(size_t)b * H_ + h] = c;
        out_newh[(size_t)b * H_ + h] = v;
        nh[j] = v; s += v;
    }
    red[tid] = s; __syncthreads();
    for (int off = 128; off > 0; off >>= 1) { if (tid < off) red[tid] += red[tid + off]; __syncthreads(); }
    const float mean = red[0] * (1.0f / H_);
    __syncthreads();
    float s2 = 0.f;
#pragma unroll
    for (int j = 0; j < 4; ++j) { float d = nh[j] - mean; s2 += d * d; }
    red[tid] = s2; __syncthreads();
    for (int off = 128; off > 0; off >>= 1) { if (tid < off) red[tid] += red[tid + off]; __syncthreads(); }
    const float rstd = rsqrtf(red[0] * (1.0f / H_) + 1e-5f);
#pragma unroll
    for (int j = 0; j < 4; ++j) {
        int h = tid + j * 256;
        float v = (nh[j] - mean) * rstd * g2[h] + b2[h];
        h_norm[(size_t)b * H_ + h] = v;
        h_norm_bf[(size_t)b * H_ + h] = (__bf16)v;
    }
}

// ---- Fused Bahdanau attention: one block per batch row ----------------------
// keys[b] is staged into LDS as bf16 (64 rows at a time, padded stride),
// k = keys[b]@Ua^T computed by WMMA with LDS A-fragments, scores reduced in
// LDS, softmax in LDS, then context + residual + LN3 -> bf16.
__global__ __launch_bounds__(256)
void attn_fused(const float* __restrict__ keys, const __bf16* __restrict__ Ua_bf,
                const float* __restrict__ Ua_b, const float* __restrict__ Va,
                const float* __restrict__ Va_b, const float* __restrict__ q,
                const float* __restrict__ h_norm, const float* __restrict__ g3,
                const float* __restrict__ b3, __bf16* __restrict__ attn_bf)
{
    __shared__ __bf16 kl[64 * KLD_];   // 129 KB staged keys half (bf16, padded)
    __shared__ float sc[S_];
    __shared__ float qs[H_];
    __shared__ float vas[H_];
    __shared__ float red[256];

    const int b = blockIdx.x, tid = threadIdx.x;
    const int lane = tid & 31, wave = tid >> 5;
    const int lm = lane & 15;
    const float* kb = keys + (size_t)b * S_ * H_;

    if (tid < S_) sc[tid] = 0.f;
#pragma unroll
    for (int j = 0; j < 4; ++j) {
        int h = tid + j * 256;
        qs[h] = q[(size_t)b * H_ + h];
        vas[h] = Va[h];
    }

    v8f zero = {};
    for (int mh = 0; mh < 2; ++mh) {
        __syncthreads();   // previous half's compute done (and qs/vas visible)
        // stage 64 rows of keys[b] as bf16 into padded LDS
        const int mrow0 = mh * 64;
#pragma unroll 4
        for (int i = 0; i < 64; ++i) {
            int lin = tid + i * 256;        // float4 index within 64x1024 half
            int row = lin >> 8;             // 0..63
            int c4  = (lin & 255) * 4;      // column (multiple of 4)
            float4 f = *(const float4*)(kb + (size_t)(mrow0 + row) * H_ + c4);
            __bf16* d = kl + row * KLD_ + c4;
            d[0] = (__bf16)f.x; d[1] = (__bf16)f.y;
            d[2] = (__bf16)f.z; d[3] = (__bf16)f.w;
        }
        __syncthreads();

        float part[4][8];
#pragma unroll
        for (int mf = 0; mf < 4; ++mf)
#pragma unroll
            for (int r = 0; r < 8; ++r) part[mf][r] = 0.f;

        // each wave owns a 64x16 tile: N columns tN..tN+15 per 128-wide slab
        for (int nb = 0; nb < H_; nb += 128) {
            const int tN = nb + wave * 16;
            v8f acc[4];
#pragma unroll
            for (int i = 0; i < 4; ++i) acc[i] = zero;

            for (int k = 0; k < H_; k += 32) {
                v16bf a[4];
#pragma unroll
                for (int mf = 0; mf < 4; ++mf)
                    a[mf] = load_fragA_bf(kl + (mf * 16 + lm) * KLD_ + k, lane);
                v16bf bbf = load_fragB_bf(Ua_bf + (size_t)(tN + lm) * H_ + k, lane);
#pragma unroll
                for (int mf = 0; mf < 4; ++mf)
                    acc[mf] = __builtin_amdgcn_wmma_f32_16x16x32_bf16(
                        false, a[mf], false, bbf, (short)0, acc[mf], false, false);
            }
            // fold this 16-col slice into per-row score partials
#pragma unroll
            for (int mf = 0; mf < 4; ++mf)
#pragma unroll
                for (int r = 0; r < 8; ++r) {
                    int h = tN + lm;
                    part[mf][r] += tanhf(qs[h] + acc[mf][r] + Ua_b[h]) * vas[h];
                }
        }
#pragma unroll
        for (int mf = 0; mf < 4; ++mf)
#pragma unroll
            for (int r = 0; r < 8; ++r) {
                int srow = mrow0 + mf * 16 + ((lane & 16) ? (8 + r) : r);
                atomicAdd(&sc[srow], part[mf][r]);
            }
    }
    __syncthreads();

    // softmax over sc[0..S_-1]
    float my = (tid < S_) ? (sc[tid] + Va_b[0]) : -3.0e38f;
    red[tid] = my; __syncthreads();
    for (int off = 128; off > 0; off >>= 1) { if (tid < off) red[tid] = fmaxf(red[tid], red[tid + off]); __syncthreads(); }
    const float mx = red[0];
    __syncthreads();
    float e = (tid < S_) ? __expf(my - mx) : 0.f;
    red[tid] = e; __syncthreads();
    for (int off = 128; off > 0; off >>= 1) { if (tid < off) red[tid] += red[tid + off]; __syncthreads(); }
    const float inv = 1.0f / red[0];
    __syncthreads();
    if (tid < S_) sc[tid] = e * inv;
    __syncthreads();

    // context (coalesced over h) + residual + LN3
    float t[4];
#pragma unroll
    for (int j = 0; j < 4; ++j) t[j] = 0.f;
    for (int s = 0; s < S_; ++s) {
        float w = sc[s];
        const float* kr = kb + (size_t)s * H_;
#pragma unroll
        for (int j = 0; j < 4; ++j) t[j] += w * kr[tid + j * 256];
    }
    float ssum = 0.f;
#pragma unroll
    for (int j = 0; j < 4; ++j) {
        t[j] += h_norm[(size_t)b * H_ + tid + j * 256];
        ssum += t[j];
    }
    red[tid] = ssum; __syncthreads();
    for (int off = 128; off > 0; off >>= 1) { if (tid < off) red[tid] += red[tid + off]; __syncthreads(); }
    const float mean = red[0] * (1.0f / H_);
    __syncthreads();
    float s2 = 0.f;
#pragma unroll
    for (int j = 0; j < 4; ++j) { float d = t[j] - mean; s2 += d * d; }
    red[tid] = s2; __syncthreads();
    for (int off = 128; off > 0; off >>= 1) { if (tid < off) red[tid] += red[tid + off]; __syncthreads(); }
    const float rstd = rsqrtf(red[0] * (1.0f / H_) + 1e-5f);
#pragma unroll
    for (int j = 0; j < 4; ++j) {
        int h = tid + j * 256;
        attn_bf[(size_t)b * H_ + h] = (__bf16)((t[j] - mean) * rstd * g3[h] + b3[h]);
    }
}

// ---- launcher ---------------------------------------------------------------
extern "C" void kernel_launch(void* const* d_in, const int* in_sizes, int n_in,
                              void* d_out, int out_size, void* d_ws, size_t ws_size,
                              hipStream_t stream)
{
    const float* x    = (const float*)d_in[0];
    const float* h0   = (const float*)d_in[1];
    const float* c0   = (const float*)d_in[2];
    const float* keys = (const float*)d_in[3];
    const float* W_ih = (const float*)d_in[4];
    const float* b_ih = (const float*)d_in[5];
    const float* W_hh = (const float*)d_in[6];
    const float* b_hh = (const float*)d_in[7];
    const float* ln1_g = (const float*)d_in[8];
    const float* ln1_b = (const float*)d_in[9];
    const float* ln2_g = (const float*)d_in[10];
    const float* ln2_b = (const float*)d_in[11];
    const float* ln3_g = (const float*)d_in[12];
    const float* ln3_b = (const float*)d_in[13];
    const float* Wa    = (const float*)d_in[14];
    const float* Wa_b  = (const float*)d_in[15];
    const float* Ua    = (const float*)d_in[16];
    const float* Ua_b  = (const float*)d_in[17];
    const float* Va    = (const float*)d_in[18];
    const float* Va_b  = (const float*)d_in[19];
    const float* w1    = (const float*)d_in[20];
    const float* w2    = (const float*)d_in[21];
    const float* w3    = (const float*)d_in[22];

    float* out   = (float*)d_out;                 // [B,H] final output
    float* out_h = out + (size_t)B_ * H_;         // new_h
    float* out_c = out + 2 * (size_t)B_ * H_;     // c_new

    // f32 workspace
    float* ws    = (float*)d_ws;
    float* gates = ws;                            // B*4H
    float* hnorm = gates + (size_t)B_ * 4 * H_;   // B*H
    float* qbuf  = hnorm + (size_t)B_ * H_;       // B*H
    float* g1    = qbuf  + (size_t)B_ * H_;       // B*F
    // bf16 workspace (after f32 region)
    __bf16* bfw     = (__bf16*)(g1 + (size_t)B_ * F_);
    __bf16* xnormb  = bfw;                        // B*H
    __bf16* h0b     = xnormb + (size_t)B_ * H_;   // B*H
    __bf16* hnormb  = h0b    + (size_t)B_ * H_;   // B*H
    __bf16* attnb   = hnormb + (size_t)B_ * H_;   // B*H
    __bf16* g1b     = attnb  + (size_t)B_ * H_;   // B*F
    __bf16* W_ihb   = g1b    + (size_t)B_ * F_;   // 4H*H
    __bf16* W_hhb   = W_ihb  + (size_t)4 * H_ * H_;
    __bf16* Wab     = W_hhb  + (size_t)4 * H_ * H_;
    __bf16* Uab     = Wab    + (size_t)H_ * H_;
    __bf16* w1b     = Uab    + (size_t)H_ * H_;
    __bf16* w2b     = w1b    + (size_t)F_ * H_;
    __bf16* w3b     = w2b    + (size_t)H_ * F_;

    // 0) one-pass bf16 conversions (weights + h0)
    cvt_bf16<<<(4 * H_ * H_) / 1024, 256, 0, stream>>>(W_ih, W_ihb);
    cvt_bf16<<<(4 * H_ * H_) / 1024, 256, 0, stream>>>(W_hh, W_hhb);
    cvt_bf16<<<(H_ * H_) / 1024, 256, 0, stream>>>(Wa, Wab);
    cvt_bf16<<<(H_ * H_) / 1024, 256, 0, stream>>>(Ua, Uab);
    cvt_bf16<<<(F_ * H_) / 1024, 256, 0, stream>>>(w1, w1b);
    cvt_bf16<<<(H_ * F_) / 1024, 256, 0, stream>>>(w2, w2b);
    cvt_bf16<<<(F_ * H_) / 1024, 256, 0, stream>>>(w3, w3b);
    cvt_bf16<<<(B_ * H_) / 1024, 256, 0, stream>>>(h0, h0b);

    // 1) x_norm = LN1(x) -> bf16
    ln_rows_bf<<<B_, 256, 0, stream>>>(x, ln1_g, ln1_b, xnormb);

    // 2) gates = x_norm@W_ih^T + h0@W_hh^T + b_ih + b_hh   (dual-pair GEMM)
    gemm_bf16_wmma<0><<<dim3(4 * H_ / 128, B_ / 128), 256, 0, stream>>>(
        xnormb, W_ihb, h0b, W_hhb, b_ih, b_hh, nullptr, gates, nullptr,
        B_, 4 * H_, H_);

    // 3) LSTM cell + residual (-> d_out new_h, c_new) + LN2 -> h_norm (f32+bf16)
    lstm_ln2<<<B_, 256, 0, stream>>>(gates, x, c0, ln2_g, ln2_b, out_h, out_c,
                                     hnorm, hnormb);

    // 4) q = h_norm@Wa^T + Wa_b
    gemm_bf16_wmma<0><<<dim3(H_ / 128, B_ / 128), 256, 0, stream>>>(
        hnormb, Wab, nullptr, nullptr, Wa_b, nullptr, nullptr, qbuf, nullptr,
        B_, H_, H_);

    // 5) fused additive attention + residual + LN3 -> attn (bf16)
    attn_fused<<<B_, 256, 0, stream>>>(keys, Uab, Ua_b, Va, Va_b, qbuf, hnorm,
                                       ln3_g, ln3_b, attnb);

    // 6) g1 = attn@w1^T   (f32)
    gemm_bf16_wmma<0><<<dim3(F_ / 128, B_ / 128), 256, 0, stream>>>(
        attnb, w1b, nullptr, nullptr, nullptr, nullptr, nullptr, g1, nullptr,
        B_, F_, H_);

    // 7) g1b = silu(g1) * (attn@w3^T)   (SwiGLU combine, bf16 out)
    gemm_bf16_wmma<1><<<dim3(F_ / 128, B_ / 128), 256, 0, stream>>>(
        attnb, w3b, nullptr, nullptr, nullptr, nullptr, g1, nullptr, g1b,
        B_, F_, H_);

    // 8) out = g1b@w2^T + new_h
    gemm_bf16_wmma<2><<<dim3(H_ / 128, B_ / 128), 256, 0, stream>>>(
        g1b, w2b, nullptr, nullptr, nullptr, nullptr, out_h, out, nullptr,
        B_, H_, F_);
}
